// ChemGCLayer_73796128080691
// MI455X (gfx1250) — matrix operand order repaired and checked by reading, not compiled, not run on hardware
//
#include <hip/hip_runtime.h>
#include <hip/hip_bf16.h>
#include <stdint.h>

typedef __attribute__((ext_vector_type(16))) _Float16 v16h;
typedef __attribute__((ext_vector_type(8)))  _Float16 v8h;
typedef __attribute__((ext_vector_type(8)))  float    v8f;

#define NTILES 16   // 256 output columns = 16 n-tiles of 16
#define WAVES_PER_BLOCK 8
#define BSLICE (NTILES * 32 * 16)   // halves per 32-wide K slice of packed B

// ---------------------------------------------------------------------------
// Pack f32 weight matrix W[K x 256] into the B-fragment order used here (f16, contiguous k):
// index = ((kt*16 + nt)*32 + lane)*16 + h
// lane<16 : col = nt*16+lane,      K = kt*32 + h        (h = 0..15)
// lane>=16: col = nt*16+(lane-16), K = kt*32 + 16 + h
// ---------------------------------------------------------------------------
__global__ void k_pack_b(const float* __restrict__ W, _Float16* __restrict__ Bp,
                         int K) {
    int idx = blockIdx.x * blockDim.x + threadIdx.x;
    int total = K * 256;
    if (idx >= total) return;
    int h    = idx & 15;
    int lane = (idx >> 4) & 31;
    int nt   = (idx >> 9) & 15;
    int kt   = idx >> 13;
    int k    = kt * 32 + ((lane < 16) ? 0 : 16) + h;
    int col  = nt * 16 + (lane & 15);
    Bp[idx] = (_Float16)W[(size_t)k * 256 + col];
}

__global__ void k_f32_to_f16(const float* __restrict__ s, _Float16* __restrict__ d,
                             long long n) {
    long long i = (long long)blockIdx.x * blockDim.x + threadIdx.x;
    if (i < n) d[i] = (_Float16)s[i];
}

__global__ void k_i64_to_f32(const long long* __restrict__ s, float* __restrict__ d,
                             long long n) {
    long long i = (long long)blockIdx.x * blockDim.x + threadIdx.x;
    if (i < n) d[i] = (float)s[i];
}

__global__ void k_zero_f32(float* __restrict__ p, long long n) {
    long long i = (long long)blockIdx.x * blockDim.x + threadIdx.x;
    if (i < n) p[i] = 0.0f;
}

// degree count over destination nodes (f32 native atomics)
__global__ void k_deg_count(const long long* __restrict__ edges, long long E,
                            float* __restrict__ deg) {
    long long e = (long long)blockIdx.x * blockDim.x + threadIdx.x;
    if (e >= E) return;
    long long dst = edges[E + e];
    atomicAdd(&deg[dst], 1.0f);
}

__global__ void k_deg_fin(float* __restrict__ deg, float* __restrict__ dinv,
                          float* __restrict__ invdeg, long long n) {
    long long i = (long long)blockIdx.x * blockDim.x + threadIdx.x;
    if (i >= n) return;
    float d = deg[i] + 1.0f;           // self loop
    deg[i] = d;
    dinv[i]   = rsqrtf(d);
    invdeg[i] = 1.0f / d;
}

// agg[i,:] = xw[i,:] * (1/deg[i]) + bg[:]
__global__ void k_agg_init(const float* __restrict__ xw, const float* __restrict__ invdeg,
                           const float* __restrict__ bg, float* __restrict__ agg,
                           long long n256) {
    long long i = (long long)blockIdx.x * blockDim.x + threadIdx.x;
    if (i >= n256) return;
    long long row = i >> 8;
    int c = (int)(i & 255);
    agg[i] = xw[i] * invdeg[row] + bg[c];
}

// one wave per edge: agg[dst,:] += xw[src,:] * dinv[src]*dinv[dst]
__global__ __launch_bounds__(256)
void k_edges(const long long* __restrict__ edges, long long E,
             const float* __restrict__ xw, const float* __restrict__ dinv,
             float* __restrict__ agg) {
    long long e = (long long)blockIdx.x * WAVES_PER_BLOCK + (threadIdx.x >> 5);
    int lane = threadIdx.x & 31;
    if (e >= E) return;
    long long s = edges[e];
    long long d = edges[E + e];
    float nrm = dinv[s] * dinv[d];
    const float* xs = xw  + s * 256;
    float*       ad = agg + d * 256;
#pragma unroll
    for (int i = 0; i < 8; ++i) {
        int c = lane + i * 32;
        atomicAdd(&ad[c], xs[c] * nrm);
    }
}

// out[:,256:384] = feats  (row stride 384 in d_out)
__global__ void k_out_feats(const float* __restrict__ feats, float* __restrict__ out,
                            long long n128) {
    long long i = (long long)blockIdx.x * blockDim.x + threadIdx.x;
    if (i >= n128) return;
    long long row = i >> 7;
    int c = (int)(i & 127);
    out[row * 384 + 256 + c] = feats[i];
}

// B fragment from staged LDS slice (32 B per lane, two ds_load_b128)
__device__ __forceinline__ v16h load_bfrag(const _Float16* base, int nt, int lane) {
    const v8h* bp = (const v8h*)(base + (nt * 32 + lane) * 16);
    v8h lo = bp[0], hi = bp[1];
    return __builtin_shufflevector(lo, hi, 0,1,2,3,4,5,6,7,8,9,10,11,12,13,14,15);
}

// A fragment (contiguous k order, same as packed B): lanes<16 K 0..15; lanes>=16 K 16..31
__device__ __forceinline__ v16h load_afrag(const _Float16* __restrict__ A0, int K0,
                                           const _Float16* __restrict__ A1, int K1,
                                           int mt, int lrow, int hsel, int kglob) {
    const _Float16* Arow;
    if (kglob < K0)
        Arow = A0 + (size_t)(mt * 16 + lrow) * K0 + kglob;
    else
        Arow = A1 + (size_t)(mt * 16 + lrow) * K1 + (kglob - K0);
    const _Float16* ap = Arow + hsel * 16;
    v8h alo = *(const v8h*)(ap);
    v8h ahi = *(const v8h*)(ap + 8);
    return __builtin_shufflevector(alo, ahi, 0,1,2,3,4,5,6,7,8,9,10,11,12,13,14,15);
}

// ---------------------------------------------------------------------------
// WMMA GEMM: C[M x 256] = [A0 | A1] (f16, row-major, K = K0+K1) x Bpacked
// Each wave: one 16-row tile x all 256 cols (16 WMMA accumulators).
// Double-buffered LDS staging of the packed B slice (2 x 16 KB).
// B fragments: 3-slot rotating pipeline + sched_group_barrier pattern
//   DS(6), 13 x [WMMA(1), DS(2)], WMMA(3)
// so each ds_load pair is issued ~2 WMMAs ahead of its consuming wait.
// ---------------------------------------------------------------------------
__global__ __launch_bounds__(256)
void k_gemm(const _Float16* __restrict__ A0, int K0,
            const _Float16* __restrict__ A1, int K1,
            const _Float16* __restrict__ Bp,
            const float* __restrict__ bias,
            float* __restrict__ outF, int outStride,
            _Float16* __restrict__ outH,
            int Mtiles, int applyElu) {
    __shared__ _Float16 Bs[2][BSLICE];   // 2 x 16 KB

    const int tid  = threadIdx.x;
    const int lane = tid & 31;
    const int w    = tid >> 5;
    const int mt   = blockIdx.x * WAVES_PER_BLOCK + w;
    const bool valid = (mt < Mtiles);
    const int KT = (K0 + K1) >> 5;
    const int lrow = lane & 15;
    const int hsel = (lane < 16) ? 0 : 1;

    v8f acc[NTILES];
#pragma unroll
    for (int nt = 0; nt < NTILES; ++nt) {
        float bv = bias ? bias[nt * 16 + lrow] : 0.0f;
        v8f a = {bv, bv, bv, bv, bv, bv, bv, bv};
        acc[nt] = a;
    }

    // prologue: global loads for B slice 0 and A fragment 0
    v8h st0, st1, st2, st3;
    {
        const v8h* src = (const v8h*)Bp + tid * 4;
        st0 = src[0]; st1 = src[1]; st2 = src[2]; st3 = src[3];
    }
    v16h afrag;
    if (valid) afrag = load_afrag(A0, K0, A1, K1, mt, lrow, hsel, 0);

    for (int kt = 0; kt < KT; ++kt) {
        {   // commit staged slice kt into its LDS buffer (64 B per thread)
            v8h* dst = ((v8h*)Bs[kt & 1]) + tid * 4;
            dst[0] = st0; dst[1] = st1; dst[2] = st2; dst[3] = st3;
        }
        __syncthreads();

        // issue next-slice global loads early; waited on at next ds_store
        if (kt + 1 < KT) {
            const v8h* src = (const v8h*)(Bp + (size_t)(kt + 1) * BSLICE) + tid * 4;
            st0 = src[0]; st1 = src[1]; st2 = src[2]; st3 = src[3];
        }

        if (valid) {
            // preload next A fragment; its wait lands in the next iteration
            v16h afrag_next;
            if (kt + 1 < KT)
                afrag_next = load_afrag(A0, K0, A1, K1, mt, lrow, hsel, (kt + 1) * 32);

            const _Float16* base = Bs[kt & 1];
            // 3-slot rotating B pipeline
            v16h b0 = load_bfrag(base, 0, lane);
            v16h b1 = load_bfrag(base, 1, lane);
            v16h b2 = load_bfrag(base, 2, lane);
#pragma unroll
            for (int nt = 0; nt < NTILES; nt += 3) {
                acc[nt] = __builtin_amdgcn_wmma_f32_16x16x32_f16(
                    false, afrag, false, b0, (short)0, acc[nt], false, false);
                if (nt + 3 < NTILES) b0 = load_bfrag(base, nt + 3, lane);
                if (nt + 1 < NTILES) {
                    acc[nt + 1] = __builtin_amdgcn_wmma_f32_16x16x32_f16(
                        false, afrag, false, b1, (short)0, acc[nt + 1], false, false);
                    if (nt + 4 < NTILES) b1 = load_bfrag(base, nt + 4, lane);
                }
                if (nt + 2 < NTILES) {
                    acc[nt + 2] = __builtin_amdgcn_wmma_f32_16x16x32_f16(
                        false, afrag, false, b2, (short)0, acc[nt + 2], false, false);
                    if (nt + 5 < NTILES) b2 = load_bfrag(base, nt + 5, lane);
                }
            }
            afrag = afrag_next;

            // Pin the issue order: 3 DS pairs up front, then WMMA/DS(2)
            // alternation, tail of 3 WMMAs. 0x100 = DS read, 0x008 = MFMA/WMMA.
            __builtin_amdgcn_sched_group_barrier(0x100, 6, 0);
#pragma unroll
            for (int i = 0; i < 13; ++i) {
                __builtin_amdgcn_sched_group_barrier(0x008, 1, 0);
                __builtin_amdgcn_sched_group_barrier(0x100, 2, 0);
            }
            __builtin_amdgcn_sched_group_barrier(0x008, 1, 0);
            __builtin_amdgcn_sched_group_barrier(0x008, 1, 0);
            __builtin_amdgcn_sched_group_barrier(0x008, 1, 0);
        }
    }

    if (valid) {
        // C/D layout: VGPR r -> M = r (lanes 0-15) / M = r+8 (lanes 16-31), N = lane%16
        const int rbase = mt * 16 + ((lane < 16) ? 0 : 8);
#pragma unroll
        for (int nt = 0; nt < NTILES; ++nt) {
            int col = nt * 16 + lrow;
#pragma unroll
            for (int r = 0; r < 8; ++r) {
                float v = acc[nt][r];
                if (applyElu) v = (v > 0.0f) ? v : (expf(v) - 1.0f);
                int row = rbase + r;
                if (outF) outF[(size_t)row * outStride + col] = v;
                if (outH) outH[(size_t)row * 256 + col] = (_Float16)v;
            }
        }
    }
}

// ---------------------------------------------------------------------------
extern "C" void kernel_launch(void* const* d_in, const int* in_sizes, int n_in,
                              void* d_out, int out_size, void* d_ws, size_t ws_size,
                              hipStream_t stream) {
    const float*     feats = (const float*)d_in[0];
    const long long* edges = (const long long*)d_in[1];
    const long long* batch = (const long long*)d_in[2];
    const float*     W1    = (const float*)d_in[3];
    const float*     b1    = (const float*)d_in[4];
    const float*     Wg    = (const float*)d_in[5];
    const float*     bg    = (const float*)d_in[6];
    const float*     Wc    = (const float*)d_in[7];
    const float*     bc    = (const float*)d_in[8];

    const long long N = in_sizes[0] / 128;
    const long long E = in_sizes[1] / 2;
    const int Mtiles = (int)(N / 16);          // N = 100000 -> 6250 tiles

    // workspace carve (256B aligned)
    uintptr_t p = (uintptr_t)d_ws;
    auto take = [&](size_t bytes) -> void* {
        void* r = (void*)p;
        p += (bytes + 255) & ~(size_t)255;
        return r;
    };
    _Float16* feats_h  = (_Float16*)take((size_t)N * 128 * 2);
    _Float16* nfeats_h = (_Float16*)take((size_t)N * 256 * 2);
    _Float16* agg_h    = (_Float16*)take((size_t)N * 256 * 2);
    float*    xw       = (float*)take((size_t)N * 256 * 4);
    float*    agg      = (float*)take((size_t)N * 256 * 4);
    float*    deg      = (float*)take((size_t)N * 4);
    float*    dinv     = (float*)take((size_t)N * 4);
    float*    invdeg   = (float*)take((size_t)N * 4);
    _Float16* W1p      = (_Float16*)take((size_t)128 * 256 * 2);
    _Float16* Wgp      = (_Float16*)take((size_t)384 * 256 * 2);
    _Float16* Wcp      = (_Float16*)take((size_t)512 * 256 * 2);

    float* out = (float*)d_out;

    const int T = 256;
    auto gb = [&](long long n) { return (unsigned)((n + T - 1) / T); };

    // 1) pack weights into WMMA B-fragment order (f16)
    k_pack_b<<<gb(128LL * 256), T, 0, stream>>>(W1, W1p, 128);
    k_pack_b<<<gb(384LL * 256), T, 0, stream>>>(Wg, Wgp, 384);
    k_pack_b<<<gb(512LL * 256), T, 0, stream>>>(Wc, Wcp, 512);

    // 2) feats -> f16
    k_f32_to_f16<<<gb(N * 128), T, 0, stream>>>(feats, feats_h, N * 128);

    // 3) degrees
    k_zero_f32<<<gb(N), T, 0, stream>>>(deg, N);
    k_deg_count<<<gb(E), T, 0, stream>>>(edges, E, deg);
    k_deg_fin<<<gb(N), T, 0, stream>>>(deg, dinv, invdeg, N);

    const unsigned gemmGrid = (unsigned)((Mtiles + WAVES_PER_BLOCK - 1) / WAVES_PER_BLOCK);

    // 4) GEMM1: nfeats = elu(feats @ W1 + b1) -> f16
    k_gemm<<<gemmGrid, T, 0, stream>>>(feats_h, 128, (const _Float16*)nullptr, 0,
                                       W1p, b1, (float*)nullptr, 0, nfeats_h,
                                       Mtiles, 1);

    // 5) GEMM2: xw = [nfeats|feats] @ Wg -> f32 (no bias/act)
    k_gemm<<<gemmGrid, T, 0, stream>>>(nfeats_h, 256, feats_h, 128,
                                       Wgp, (const float*)nullptr, xw, 256,
                                       (_Float16*)nullptr, Mtiles, 0);

    // 6) aggregation: agg = xw/deg + bg, then edge scatter-add (L2-resident)
    k_agg_init<<<gb(N * 256), T, 0, stream>>>(xw, invdeg, bg, agg, N * 256);
    k_edges<<<(unsigned)((E + WAVES_PER_BLOCK - 1) / WAVES_PER_BLOCK), T, 0, stream>>>(
        edges, E, xw, dinv, agg);
    k_f32_to_f16<<<gb(N * 256), T, 0, stream>>>(agg, agg_h, N * 256);

    // 7) GEMM3: out[:, :256] = elu([nfeats|agg] @ Wc + bc), row stride 384
    k_gemm<<<gemmGrid, T, 0, stream>>>(nfeats_h, 256, agg_h, 256,
                                       Wcp, bc, out, 384, (_Float16*)nullptr,
                                       Mtiles, 1);

    // 8) out[:, 256:384] = feats ; tail = edges, batch as f32
    k_out_feats<<<gb(N * 128), T, 0, stream>>>(feats, out, N * 128);
    float* tail = out + N * 384;
    k_i64_to_f32<<<gb(2 * E), T, 0, stream>>>(edges, tail, 2 * E);
    k_i64_to_f32<<<gb(N), T, 0, stream>>>(batch, tail + 2 * E, N);
}
